// CoreAttention_35493609734503
// MI455X (gfx1250) — matrix operand order repaired and checked by reading, not compiled, not run on hardware
//
#include <hip/hip_runtime.h>
#include <hip/hip_bf16.h>
#include <cstdint>

// ---- CDNA5 (gfx1250) vector types for WMMA ----
typedef __attribute__((ext_vector_type(16))) __bf16 v16bf;
typedef __attribute__((ext_vector_type(8)))  __bf16 bf16x8;
typedef __attribute__((ext_vector_type(8)))  float  v8f;
typedef __attribute__((ext_vector_type(4)))  unsigned int u32x4;
typedef __attribute__((ext_vector_type(4)))  int i32x4;
typedef __attribute__((ext_vector_type(8)))  int i32x8;

union BF16x16 { v16bf v; bf16x8 h[2]; };

#define HEADS   32
#define KVHEADS 2
#define QLEN    2048
#define KVLEN   2048
#define DHEAD   128
#define BQ      64    // query rows per block (16 per wave x 4 waves)
#define KT      32    // kv keys per iteration
#define WAVES   4
#define SCALE   0.08838834764831845f   // coeff / (sqrt(D)*coeff) = 1/sqrt(128)

struct __align__(16) Smem {
  float  mask[BQ * KT];          // 8 KB, filled by TDM
  __bf16 Kt[KT * DHEAD];         // 8 KB, bf16 row-major [key][dim]
  __bf16 Vt[DHEAD * KT];         // 8 KB, bf16 transposed [dim][key]
  __bf16 P[WAVES][16 * KT];      // 4 KB, per-wave P bounce (C->A relayout)
};

__global__ __launch_bounds__(128, 1)
void fattn_gqa_bf16_wmma(const float* __restrict__ Qp,
                         const float* __restrict__ Kp,
                         const float* __restrict__ Vp,
                         const float* __restrict__ Mp,
                         float* __restrict__ Op)
{
  __shared__ Smem s;
  const int lane = threadIdx.x & 31;
  const int wave = threadIdx.x >> 5;
  const int q0 = blockIdx.x * BQ;
  const int h  = blockIdx.y;
  const int b  = blockIdx.z;
  const int Bn = gridDim.z;
  const int g  = h / (HEADS / KVHEADS);

  const float* Qb = Qp + ((size_t)(b * HEADS + h) * QLEN + q0) * DHEAD;
  const float* Kb = Kp + (size_t)(b * KVHEADS + g) * KVLEN * DHEAD;
  const float* Vb = Vp + (size_t)(b * KVHEADS + g) * KVLEN * DHEAD;
  const float* Mb = Mp + (size_t)b * QLEN * KVLEN + (size_t)q0 * KVLEN;

  const int m16   = lane & 15;            // row (A/B column) selector
  const int kbA   = (lane < 16) ? 0 : 8;  // A-layout K-group base
  const int kbB   = (lane < 16) ? 0 : 16; // unused; B operands use kbA like A
  const int rowHi = (lane >= 16) ? 8 : 0; // C-layout row offset

  // ---- Load this wave's 16x128 Q tile into 4 bf16 A-operands (registers) ----
  v16bf qa[4];
  {
    const float* qrow = Qb + (size_t)(wave * 16 + m16) * DHEAD;
#pragma unroll
    for (int c = 0; c < 4; ++c) {
      const float4 a0 = ((const float4*)(qrow + 32 * c + kbA))[0];
      const float4 a1 = ((const float4*)(qrow + 32 * c + kbA))[1];
      const float4 b0 = ((const float4*)(qrow + 32 * c + kbA + 16))[0];
      const float4 b1 = ((const float4*)(qrow + 32 * c + kbA + 16))[1];
      qa[c][0]=(__bf16)a0.x; qa[c][1]=(__bf16)a0.y; qa[c][2]=(__bf16)a0.z; qa[c][3]=(__bf16)a0.w;
      qa[c][4]=(__bf16)a1.x; qa[c][5]=(__bf16)a1.y; qa[c][6]=(__bf16)a1.z; qa[c][7]=(__bf16)a1.w;
      qa[c][8]=(__bf16)b0.x; qa[c][9]=(__bf16)b0.y; qa[c][10]=(__bf16)b0.z; qa[c][11]=(__bf16)b0.w;
      qa[c][12]=(__bf16)b1.x; qa[c][13]=(__bf16)b1.y; qa[c][14]=(__bf16)b1.z; qa[c][15]=(__bf16)b1.w;
    }
  }

  v8f o[8];
#pragma unroll
  for (int t = 0; t < 8; ++t) o[t] = (v8f){0.f,0.f,0.f,0.f,0.f,0.f,0.f,0.f};
  float mrow[8], lrow[8];
#pragma unroll
  for (int v = 0; v < 8; ++v) { mrow[v] = -3.0e38f; lrow[v] = 0.f; }

  const uint32_t maskLdsOff = (uint32_t)(uintptr_t)(&s.mask[0]);

  for (int kv0 = 0; kv0 < KVLEN; kv0 += KT) {
    // ---- Mask tile via Tensor Data Mover (wave 0 issues, TENSORcnt-tracked) ----
#if __has_builtin(__builtin_amdgcn_tensor_load_to_lds)
    if (wave == 0) {
      const uint64_t ga = (uint64_t)(uintptr_t)(Mb + kv0);
      u32x4 g0;
      g0[0] = 1u;                                   // count=1, no gather
      g0[1] = maskLdsOff;                           // lds_addr
      g0[2] = (uint32_t)ga;                         // global_addr[31:0]
      g0[3] = ((uint32_t)(ga >> 32) & 0x01FFFFFFu) | 0x80000000u; // addr[56:32] | type=2
      i32x8 g1;
      g1[0] = (int)(2u << 16);                      // data_size=4B, wg_mask=0
      g1[1] = (int)((uint32_t)(KVLEN & 0xFFFF) << 16); // tensor_dim0[15:0]
      g1[2] = (int)((uint32_t)(QLEN  & 0xFFFF) << 16); // dim0 hi=0 | tensor_dim1[15:0]
      g1[3] = (int)((uint32_t)KT << 16);            // dim1 hi=0 | tile_dim0=32
      g1[4] = BQ;                                   // tile_dim1=64, tile_dim2=0
      g1[5] = KVLEN;                                // tensor_dim0_stride[31:0]
      g1[6] = 0;                                    // stride hi | dim1_stride lo
      g1[7] = 0;
      i32x4 z4; z4[0]=0; z4[1]=0; z4[2]=0; z4[3]=0;
      i32x8 z8; z8[0]=0; z8[1]=0; z8[2]=0; z8[3]=0; z8[4]=0; z8[5]=0; z8[6]=0; z8[7]=0;
      __builtin_amdgcn_tensor_load_to_lds(g0, g1, z4, z4, z8, 0);
    }
#else
    {
      const int tid = threadIdx.x;
#pragma unroll
      for (int i = 0; i < 16; ++i) {
        int idx = tid * 16 + i, r = idx >> 5, c = idx & 31;
        s.mask[idx] = Mb[(size_t)r * KVLEN + kv0 + c];
      }
    }
#endif

    // ---- Cooperative K/V staging: fp32 HBM -> bf16 LDS (V transposed) ----
    {
      const int row  = threadIdx.x >> 2;        // key 0..31
      const int dseg = (threadIdx.x & 3) * 32;  // dim segment
      const float* kr = Kb + (size_t)(kv0 + row) * DHEAD + dseg;
      const float* vr = Vb + (size_t)(kv0 + row) * DHEAD + dseg;
#pragma unroll
      for (int i = 0; i < 4; ++i) {
        const float4 x = ((const float4*)kr)[2 * i];
        const float4 y = ((const float4*)kr)[2 * i + 1];
        bf16x8 pk;
        pk[0]=(__bf16)x.x; pk[1]=(__bf16)x.y; pk[2]=(__bf16)x.z; pk[3]=(__bf16)x.w;
        pk[4]=(__bf16)y.x; pk[5]=(__bf16)y.y; pk[6]=(__bf16)y.z; pk[7]=(__bf16)y.w;
        *(bf16x8*)&s.Kt[row * DHEAD + dseg + 8 * i] = pk;
      }
#pragma unroll
      for (int i = 0; i < 8; ++i) {
        const float4 x = ((const float4*)vr)[i];
        const int d = dseg + 4 * i;
        s.Vt[(d + 0) * KT + row] = (__bf16)x.x;
        s.Vt[(d + 1) * KT + row] = (__bf16)x.y;
        s.Vt[(d + 2) * KT + row] = (__bf16)x.z;
        s.Vt[(d + 3) * KT + row] = (__bf16)x.w;
      }
      if (kv0 + KT < KVLEN) {
        __builtin_prefetch(kr + KT * DHEAD, 0, 1);
        __builtin_prefetch(vr + KT * DHEAD, 0, 1);
      }
    }
#if __has_builtin(__builtin_amdgcn_s_wait_tensorcnt)
    if (wave == 0) __builtin_amdgcn_s_wait_tensorcnt(0);
#endif
    __syncthreads();

    // ---- S = Q K^T (bf16 WMMA, fp32 acc), + scale + mask, C-layout ----
    v8f st[2];
#pragma unroll
    for (int kvg = 0; kvg < 2; ++kvg) {
      v8f acc = (v8f){0.f,0.f,0.f,0.f,0.f,0.f,0.f,0.f};
      const int keyrow = kvg * 16 + m16;
#pragma unroll
      for (int c = 0; c < 4; ++c) {
        BF16x16 bb;
        const __bf16* kp = &s.Kt[keyrow * DHEAD + 32 * c + kbA];
        bb.h[0] = *(const bf16x8*)kp;
        bb.h[1] = *(const bf16x8*)(kp + 16);
        acc = __builtin_amdgcn_wmma_f32_16x16x32_bf16(false, qa[c], false, bb.v,
                                                      (short)0, acc, false, false);
      }
#pragma unroll
      for (int v = 0; v < 8; ++v)
        acc[v] = acc[v] * SCALE + s.mask[(wave * 16 + v + rowHi) * KT + kvg * 16 + m16];
      st[kvg] = acc;
    }

    // ---- Online softmax (row stats across 16-lane halves) ----
    float mnew[8], al[8];
#pragma unroll
    for (int v = 0; v < 8; ++v) {
      float mx = fmaxf(st[0][v], st[1][v]);
#pragma unroll
      for (int off = 8; off >= 1; off >>= 1) mx = fmaxf(mx, __shfl_xor(mx, off, 16));
      mnew[v] = fmaxf(mrow[v], mx);
      al[v]   = __expf(mrow[v] - mnew[v]);
      mrow[v] = mnew[v];
    }
#pragma unroll
    for (int v = 0; v < 8; ++v) {
      const float p0 = __expf(st[0][v] - mnew[v]);
      const float p1 = __expf(st[1][v] - mnew[v]);
      st[0][v] = p0; st[1][v] = p1;
      float sum = p0 + p1;
#pragma unroll
      for (int off = 8; off >= 1; off >>= 1) sum += __shfl_xor(sum, off, 16);
      lrow[v] = lrow[v] * al[v] + sum;
    }

    // ---- P (C-layout fp32) -> bf16 LDS bounce -> A-layout operand ----
    {
      __bf16* pw = &s.P[wave][0];
#pragma unroll
      for (int kvg = 0; kvg < 2; ++kvg)
#pragma unroll
        for (int v = 0; v < 8; ++v)
          pw[(v + rowHi) * KT + kvg * 16 + m16] = (__bf16)st[kvg][v];
    }
#pragma unroll
    for (int t = 0; t < 8; ++t)
#pragma unroll
      for (int v = 0; v < 8; ++v) o[t][v] *= al[v];

    BF16x16 pa;
    {
      const __bf16* pr = &s.P[wave][m16 * KT + kbA];
      pa.h[0] = *(const bf16x8*)pr;        // k = kbA..kbA+7
      pa.h[1] = *(const bf16x8*)(pr + 16); // k = kbA+16..kbA+23
    }

    // ---- O += P V (8 d-tiles of 16) ----
#pragma unroll
    for (int t = 0; t < 8; ++t) {
      BF16x16 bv;
      const __bf16* vp = &s.Vt[(16 * t + m16) * KT + kbA];
      bv.h[0] = *(const bf16x8*)vp;
      bv.h[1] = *(const bf16x8*)(vp + 16);
      o[t] = __builtin_amdgcn_wmma_f32_16x16x32_bf16(false, pa.v, false, bv.v,
                                                     (short)0, o[t], false, false);
    }
    __syncthreads();
  }

  // ---- Normalize and store: out[q][b][h*128 + d] ----
  float linv[8];
#pragma unroll
  for (int v = 0; v < 8; ++v) linv[v] = 1.0f / lrow[v];
  const size_t qstride = (size_t)Bn * HEADS * DHEAD;
#pragma unroll
  for (int t = 0; t < 8; ++t) {
#pragma unroll
    for (int v = 0; v < 8; ++v) {
      const int q = q0 + wave * 16 + v + rowHi;
      const size_t idx = (size_t)q * qstride + (size_t)b * HEADS * DHEAD
                       + h * DHEAD + 16 * t + m16;
      Op[idx] = o[t][v] * linv[v];
    }
  }
}

extern "C" void kernel_launch(void* const* d_in, const int* in_sizes, int n_in,
                              void* d_out, int out_size, void* d_ws, size_t ws_size,
                              hipStream_t stream) {
  (void)n_in; (void)out_size; (void)d_ws; (void)ws_size;
  const float* Q = (const float*)d_in[0];
  const float* K = (const float*)d_in[1];
  const float* V = (const float*)d_in[2];
  const float* M = (const float*)d_in[3];
  const int B = in_sizes[0] / (HEADS * QLEN * DHEAD);
  dim3 grid(QLEN / BQ, HEADS, B);
  fattn_gqa_bf16_wmma<<<grid, dim3(128), 0, stream>>>(Q, K, V, M, (float*)d_out);
}
